// ConditionalGATrBlock_16870631539429
// MI455X (gfx1250) — matrix operand order repaired and checked by reading, not compiled, not run on hardware
//
#include <hip/hip_runtime.h>

// ---------------------------------------------------------------------------
// Types
// ---------------------------------------------------------------------------
typedef _Float16 h16;
typedef __attribute__((ext_vector_type(8)))  _Float16 v8h;
typedef __attribute__((ext_vector_type(16))) _Float16 v16h;
typedef __attribute__((ext_vector_type(8)))  float    v8f;

#define B_  4
#define N_  1024
#define M_  1024
#define H_  8
#define TOK (B_ * N_)      // 4096 (same for condition: B_*M_)
#define WPB 4              // waves per block in flash kernel
#define QT  2              // query tiles (of 16 rows) per wave

// ---------------------------------------------------------------------------
// Cl(1,3) tables
// blade order: (), e0,e1,e2,e3, e01,e02,e03,e12,e13,e23, e012,e013,e023,e123, e0123
// ---------------------------------------------------------------------------
__constant__ float kMetric[16] = {  // rev(g) * sign(b*b), metric diag(+,-,-,-)
    1.f,  1.f, -1.f, -1.f, -1.f,
   -1.f, -1.f, -1.f,  1.f,  1.f,  1.f,
    1.f,  1.f,  1.f, -1.f,
   -1.f };
__constant__ int kGrade[16] = {0,1,1,1,1,2,2,2,2,2,2,3,3,3,3,4};
__constant__ int kMask[16]  = {0,1,2,4,8,3,5,9,6,10,12,7,11,13,14,15};
__constant__ int kIdxOf[16] = {0,1,2,5,3,6,8,11,4,7,9,12,10,13,14,15};

__device__ inline float geo_sign(unsigned a, unsigned b) {
  int cnt = 0;
  unsigned t = a >> 1;
  while (t) { cnt += __popc(t & b); t >>= 1; }
  cnt += __popc((a & b) & 0xEu);   // e1,e2,e3 square to -1
  return (cnt & 1) ? -1.f : 1.f;
}

__device__ inline float gelu_tanh(float x) {
  float x3 = x * x * x;
  return 0.5f * x * (1.f + tanhf(0.7978845608028654f * (x + 0.044715f * x3)));
}

__device__ inline v16h cat8(v8h lo, v8h hi8) {
  v16h r;
#pragma unroll
  for (int i = 0; i < 8; ++i) { r[i] = lo[i]; r[8 + i] = hi8[i]; }
  return r;
}

// ---------------------------------------------------------------------------
// Equivariant LayerNorm: mv [tok,16,16], s [tok,64]. One block (64 thr)/token.
// ---------------------------------------------------------------------------
__global__ __launch_bounds__(64)
void ln_kernel(const float* __restrict__ mv_in, const float* __restrict__ s_in,
               float* __restrict__ mv_out, float* __restrict__ s_out) {
  const int tok = blockIdx.x, t = threadIdx.x;
  __shared__ float sh[64];
  __shared__ float sFactor, sMu, sVar;
  const float* mv = mv_in + (size_t)tok * 256;

  float v = 0.f;
  if (t < 16) {
    float acc = 0.f;
#pragma unroll
    for (int k = 0; k < 16; ++k) { float x = mv[t*16 + k]; acc += kMetric[k] * x * x; }
    v = fabsf(acc);
  }
  sh[t] = (t < 16) ? v : 0.f;
  __syncthreads();
  if (t == 0) {
    float s = 0.f;
#pragma unroll
    for (int i = 0; i < 16; ++i) s += sh[i];
    sFactor = rsqrtf(s * (1.f / 16.f) + 1e-6f);
  }
  __syncthreads();
  const float f = sFactor;
  for (int idx = t; idx < 256; idx += 64)
    mv_out[(size_t)tok * 256 + idx] = mv[idx] * f;

  const float x = s_in[(size_t)tok * 64 + t];
  sh[t] = x; __syncthreads();
  if (t == 0) { float s = 0.f; for (int i = 0; i < 64; ++i) s += sh[i]; sMu = s * (1.f/64.f); }
  __syncthreads();
  const float mu = sMu;
  sh[t] = (x - mu) * (x - mu); __syncthreads();
  if (t == 0) { float s = 0.f; for (int i = 0; i < 64; ++i) s += sh[i]; sVar = s * (1.f/64.f); }
  __syncthreads();
  s_out[(size_t)tok * 64 + t] = (x - mu) * rsqrtf(sVar + 1e-6f);
}

// ---------------------------------------------------------------------------
// equi_linear (per-grade channel mixing + scalar<->mv couplings), 1 block/token
// ---------------------------------------------------------------------------
__global__ __launch_bounds__(128)
void equi_linear_kernel(const float* __restrict__ mv_in, const float* __restrict__ s_in,
                        const float* __restrict__ w_mv, const float* __restrict__ w_s2mv,
                        const float* __restrict__ w_s,  const float* __restrict__ w_mv2s,
                        const float* __restrict__ b_s,
                        const float* __restrict__ res_mv, const float* __restrict__ res_s,
                        float* __restrict__ out_mv, float* __restrict__ out_s,
                        int IMV, int OMV, int IS, int OS) {
  const int tok = blockIdx.x, tid = threadIdx.x;
  __shared__ float smv[512];
  __shared__ float ss[128];
  for (int idx = tid; idx < IMV * 16; idx += 128)
    smv[idx] = mv_in[(size_t)tok * IMV * 16 + idx];
  for (int idx = tid; idx < IS; idx += 128)
    ss[idx] = s_in[(size_t)tok * IS + idx];
  __syncthreads();

  for (int idx = tid; idx < OMV * 16; idx += 128) {
    int o = idx >> 4, k = idx & 15;
    float acc = 0.f;
    const float* wrow = w_mv + (size_t)o * IMV * 5 + kGrade[k];
    for (int i = 0; i < IMV; ++i) acc += smv[i*16 + k] * wrow[i*5];
    if (k == 0) {
      const float* w2 = w_s2mv + (size_t)o * IS;
      for (int s = 0; s < IS; ++s) acc += ss[s] * w2[s];
    }
    if (res_mv) acc += res_mv[(size_t)tok * OMV * 16 + idx];
    out_mv[(size_t)tok * OMV * 16 + idx] = acc;
  }
  for (int t = tid; t < OS; t += 128) {
    float acc = b_s[t];
    const float* ws = w_s + (size_t)t * IS;
    for (int s = 0; s < IS; ++s) acc += ss[s] * ws[s];
    const float* wm = w_mv2s + (size_t)t * IMV;
    for (int i = 0; i < IMV; ++i) acc += smv[i*16] * wm[i];
    if (res_s) acc += res_s[(size_t)tok * OS + t];
    out_s[(size_t)tok * OS + t] = acc;
  }
}

// ---------------------------------------------------------------------------
// Pack Q/K per-head f16 features, row-major [B,H,Ntok,96]:
//   0..63 mv (4ch x 16 blades), 64..79 scalars, 80..95 zero.
// Q gets BLADE_METRIC and 1/sqrt(80) folded in.
// ---------------------------------------------------------------------------
__global__ void pack_kernel(const float* __restrict__ mv, const float* __restrict__ s,
                            h16* __restrict__ out, int Ntok, int is_q) {
  const int idx = blockIdx.x * blockDim.x + threadIdx.x;
  const int total = B_ * H_ * Ntok * 96;
  if (idx >= total) return;
  const int f = idx % 96;
  const int r = idx / 96;               // (b*H + h)*Ntok + n
  const int n = r % Ntok;
  const int bh = r / Ntok;
  const int h = bh & (H_ - 1), b = bh >> 3;
  const float scale = is_q ? 0.11180339887498949f : 1.f;  // 1/sqrt(80)
  float val = 0.f;
  if (f < 64) {
    const int c = f >> 4, k = f & 15;
    val = mv[(((size_t)b * Ntok + n) * 32 + h * 4 + c) * 16 + k];
    if (is_q) val *= kMetric[k];
  } else if (f < 80) {
    val = s[((size_t)b * Ntok + n) * 128 + h * 16 + (f - 64)];
  }
  out[idx] = (h16)(val * scale);
}

// Pack V feature-major [B,H,96,Ntok] so P@V B-fragments are contiguous per lane.
__global__ void pack_v_kernel(const float* __restrict__ mv, const float* __restrict__ s,
                              h16* __restrict__ out, int Ntok) {
  const int idx = blockIdx.x * blockDim.x + threadIdx.x;
  const int total = B_ * H_ * 96 * Ntok;
  if (idx >= total) return;
  const int n = idx % Ntok;
  const int r = idx / Ntok;             // (b*H + h)*96 + f
  const int f = r % 96;
  const int bh = r / 96;
  const int h = bh & (H_ - 1), b = bh >> 3;
  float val = 0.f;
  if (f < 64) {
    val = mv[(((size_t)b * Ntok + n) * 32 + h * 4 + (f >> 4)) * 16 + (f & 15)];
  } else if (f < 80) {
    val = s[((size_t)b * Ntok + n) * 128 + h * 16 + (f - 64)];
  }
  out[idx] = (h16)val;
}

// ---------------------------------------------------------------------------
// online softmax for one query tile over a 32-wide KV slab; writes f16 P tile
// ---------------------------------------------------------------------------
__device__ inline void online_softmax(v8f& c0, v8f& c1, float* m_r, float* al,
                                      h16* sP, int mrow, int hi) {
#pragma unroll
  for (int r = 0; r < 8; ++r) {
    float mloc = fmaxf(c0[r], c1[r]);
    mloc = fmaxf(mloc, __shfl_xor(mloc, 1));
    mloc = fmaxf(mloc, __shfl_xor(mloc, 2));
    mloc = fmaxf(mloc, __shfl_xor(mloc, 4));
    mloc = fmaxf(mloc, __shfl_xor(mloc, 8));
    const float mn = fmaxf(m_r[r], mloc);
    al[r] = __expf(m_r[r] - mn);
    m_r[r] = mn;
    c0[r] = __expf(c0[r] - mn);
    c1[r] = __expf(c1[r] - mn);
    sP[(r + 8 * hi) * 32 + mrow]      = (h16)c0[r];
    sP[(r + 8 * hi) * 32 + 16 + mrow] = (h16)c1[r];
  }
}

// ---------------------------------------------------------------------------
// Flash attention: one wave owns 2 query tiles (32 rows); KV slab = 32 rows.
// Per slab: 12 S-WMMA + 2 rowsum-WMMA + 10 O-WMMA, all operands loaded from
// global as contiguous b128 pairs before each WMMA burst (overlap + clauses).
// Only the 16x32 P tiles round-trip through wave-private LDS.
// ---------------------------------------------------------------------------
__global__ __launch_bounds__(32 * WPB)
void flash_attn_kernel(const h16* __restrict__ qh,   // [bh][N_][96]
                       const h16* __restrict__ kh,   // [bh][M_][96]
                       const h16* __restrict__ vt,   // [bh][96][M_]
                       float* __restrict__ o_mv, float* __restrict__ o_s) {
  const int lane = threadIdx.x & 31, wv = threadIdx.x >> 5;
  const int bh = blockIdx.y, b = bh >> 3, h = bh & (H_ - 1);
  const int n0 = (blockIdx.x * WPB + wv) * (16 * QT);
  const int mrow = lane & 15, hi = lane >> 4;

  __shared__ __align__(16) h16 sPall[WPB * QT * 16 * 32];
  h16* sP0 = sPall + wv * QT * 16 * 32;
  h16* sP1 = sP0 + 16 * 32;

  // Q fragments for both tiles (A-layout: two contiguous 16B runs per chunk)
  v16h aq0[3], aq1[3];
  {
    const h16* q0 = qh + ((size_t)bh * N_ + n0 + mrow) * 96;
    const h16* q1 = q0 + 16 * 96;
#pragma unroll
    for (int kk = 0; kk < 3; ++kk) {
      aq0[kk] = cat8(*(const v8h*)(q0 + kk * 32 + 8 * hi),
                     *(const v8h*)(q0 + kk * 32 + 16 + 8 * hi));
      aq1[kk] = cat8(*(const v8h*)(q1 + kk * 32 + 8 * hi),
                     *(const v8h*)(q1 + kk * 32 + 16 + 8 * hi));
    }
  }
  v16h bones;
#pragma unroll
  for (int i = 0; i < 16; ++i) bones[i] = (h16)1.f;

  const v8f vzero = {0.f,0.f,0.f,0.f,0.f,0.f,0.f,0.f};
  float m0[8], l0[8], a0[8], m1[8], l1[8], a1[8];
  v8f acc0[5], acc1[5];
#pragma unroll
  for (int r = 0; r < 8; ++r) { m0[r] = m1[r] = -1e30f; l0[r] = l1[r] = 0.f; }
#pragma unroll
  for (int t = 0; t < 5; ++t) { acc0[t] = vzero; acc1[t] = vzero; }

  const h16* kw = kh + (size_t)bh * M_ * 96 + (size_t)mrow * 96 + 8 * hi;
  const h16* vw = vt + ((size_t)bh * 96 + mrow) * M_ + 8 * hi;

  for (int mt = 0; mt < M_ / 32; ++mt) {
    const h16* kptr = kw + (size_t)mt * 32 * 96;
    const h16* vptr = vw + mt * 32;
    if (mt + 1 < M_ / 32) {
      __builtin_prefetch(kptr + 32 * 96, 0, 3);
      __builtin_prefetch(vptr + 32, 0, 3);
    }

    // ---- load ALL K fragments first (12 x b128, clause-able), then 12 WMMAs
    v16h bk00, bk01, bk02, bk10, bk11, bk12;
    bk00 = cat8(*(const v8h*)(kptr +          0), *(const v8h*)(kptr +         16));
    bk01 = cat8(*(const v8h*)(kptr +         32), *(const v8h*)(kptr +         48));
    bk02 = cat8(*(const v8h*)(kptr +         64), *(const v8h*)(kptr +         80));
    bk10 = cat8(*(const v8h*)(kptr + 16*96 +  0), *(const v8h*)(kptr + 16*96 + 16));
    bk11 = cat8(*(const v8h*)(kptr + 16*96 + 32), *(const v8h*)(kptr + 16*96 + 48));
    bk12 = cat8(*(const v8h*)(kptr + 16*96 + 64), *(const v8h*)(kptr + 16*96 + 80));

    v8f s00 = vzero, s01 = vzero, s10 = vzero, s11 = vzero;  // [qt][kv-subtile]
    s00 = __builtin_amdgcn_wmma_f32_16x16x32_f16(false, aq0[0], false, bk00, (short)0, s00, false, false);
    s01 = __builtin_amdgcn_wmma_f32_16x16x32_f16(false, aq0[0], false, bk10, (short)0, s01, false, false);
    s10 = __builtin_amdgcn_wmma_f32_16x16x32_f16(false, aq1[0], false, bk00, (short)0, s10, false, false);
    s11 = __builtin_amdgcn_wmma_f32_16x16x32_f16(false, aq1[0], false, bk10, (short)0, s11, false, false);
    s00 = __builtin_amdgcn_wmma_f32_16x16x32_f16(false, aq0[1], false, bk01, (short)0, s00, false, false);
    s01 = __builtin_amdgcn_wmma_f32_16x16x32_f16(false, aq0[1], false, bk11, (short)0, s01, false, false);
    s10 = __builtin_amdgcn_wmma_f32_16x16x32_f16(false, aq1[1], false, bk01, (short)0, s10, false, false);
    s11 = __builtin_amdgcn_wmma_f32_16x16x32_f16(false, aq1[1], false, bk11, (short)0, s11, false, false);
    s00 = __builtin_amdgcn_wmma_f32_16x16x32_f16(false, aq0[2], false, bk02, (short)0, s00, false, false);
    s01 = __builtin_amdgcn_wmma_f32_16x16x32_f16(false, aq0[2], false, bk12, (short)0, s01, false, false);
    s10 = __builtin_amdgcn_wmma_f32_16x16x32_f16(false, aq1[2], false, bk02, (short)0, s10, false, false);
    s11 = __builtin_amdgcn_wmma_f32_16x16x32_f16(false, aq1[2], false, bk12, (short)0, s11, false, false);

    // ---- online softmax for both tiles (independent -> ILP over hazards)
    online_softmax(s00, s01, m0, a0, sP0, mrow, hi);
    online_softmax(s10, s11, m1, a1, sP1, mrow, hi);

    // wave-internal DS ordering fence (CDNA5 split counters) + compiler fence
    asm volatile("s_wait_dscnt 0x0" ::: "memory");
    const v16h ap0 = cat8(*(const v8h*)(sP0 + mrow * 32 + 8 * hi),
                          *(const v8h*)(sP0 + mrow * 32 + 16 + 8 * hi));
    const v16h ap1 = cat8(*(const v8h*)(sP1 + mrow * 32 + 8 * hi),
                          *(const v8h*)(sP1 + mrow * 32 + 16 + 8 * hi));

    // ---- row sums via WMMA against ones (replicated across N)
    v8f cs0 = __builtin_amdgcn_wmma_f32_16x16x32_f16(false, ap0, false, bones, (short)0, vzero, false, false);
    v8f cs1 = __builtin_amdgcn_wmma_f32_16x16x32_f16(false, ap1, false, bones, (short)0, vzero, false, false);
#pragma unroll
    for (int r = 0; r < 8; ++r) {
      l0[r] = l0[r] * a0[r] + cs0[r];
      l1[r] = l1[r] * a1[r] + cs1[r];
    }

    // ---- load ALL V fragments (10 x b128), then O updates (10 WMMAs)
    v16h bv[5];
#pragma unroll
    for (int t = 0; t < 5; ++t)
      bv[t] = cat8(*(const v8h*)(vptr + (size_t)t * 16 * M_),
                   *(const v8h*)(vptr + (size_t)t * 16 * M_ + 16));
#pragma unroll
    for (int t = 0; t < 5; ++t) {
#pragma unroll
      for (int r = 0; r < 8; ++r) { acc0[t][r] *= a0[r]; acc1[t][r] *= a1[r]; }
      acc0[t] = __builtin_amdgcn_wmma_f32_16x16x32_f16(false, ap0, false, bv[t], (short)0, acc0[t], false, false);
      acc1[t] = __builtin_amdgcn_wmma_f32_16x16x32_f16(false, ap1, false, bv[t], (short)0, acc1[t], false, false);
    }
    asm volatile("" ::: "memory");  // keep next iter's sP stores after ap use
  }

  // ---- epilogue: normalize, scatter into o_mv [B,N,32,16] / o_s [B,N,128]
#pragma unroll
  for (int t = 0; t < 5; ++t)
#pragma unroll
    for (int r = 0; r < 8; ++r) {
      const int m = r + 8 * hi;
      const int d = t * 16 + mrow;
      const int row0 = n0 + m, row1 = n0 + 16 + m;
      const float val0 = acc0[t][r] / l0[r];
      const float val1 = acc1[t][r] / l1[r];
      if (d < 64) {
        o_mv[(((size_t)b * N_ + row0) * 32 + h * 4 + (d >> 4)) * 16 + (d & 15)] = val0;
        o_mv[(((size_t)b * N_ + row1) * 32 + h * 4 + (d >> 4)) * 16 + (d & 15)] = val1;
      } else {
        o_s[((size_t)b * N_ + row0) * 128 + h * 16 + (d - 64)] = val0;
        o_s[((size_t)b * N_ + row1) * 128 + h * 16 + (d - 64)] = val1;
      }
    }
}

// ---------------------------------------------------------------------------
// Geometric product + scalar-gated GELU. h_mv [tok,32,16] -> g_mv [tok,16,16]
// ---------------------------------------------------------------------------
__global__ __launch_bounds__(256)
void geo_kernel(const float* __restrict__ h_mv, float* __restrict__ g_mv) {
  const int tok = blockIdx.x, t = threadIdx.x;
  __shared__ float sh[512];
  __shared__ float sg[256];
  const float* hp = h_mv + (size_t)tok * 512;
  sh[t] = hp[t]; sh[t + 256] = hp[t + 256];
  __syncthreads();
  const int c = t >> 4, k = t & 15;
  const unsigned mk = (unsigned)kMask[k];
  float g = 0.f;
#pragma unroll
  for (int i = 0; i < 16; ++i) {
    const unsigned ma = (unsigned)kMask[i];
    const unsigned mb = ma ^ mk;
    const int j = kIdxOf[mb];
    g += sh[c * 16 + i] * sh[256 + c * 16 + j] * geo_sign(ma, mb);
  }
  sg[t] = g;
  __syncthreads();
  g_mv[(size_t)tok * 256 + t] = sg[t] * gelu_tanh(sg[c * 16]);
}

__global__ void gelu_kernel(const float* __restrict__ in, float* __restrict__ out, int n) {
  const int i = blockIdx.x * blockDim.x + threadIdx.x;
  if (i < n) out[i] = gelu_tanh(in[i]);
}

// ---------------------------------------------------------------------------
// Host launch
// ---------------------------------------------------------------------------
extern "C" void kernel_launch(void* const* d_in, const int* in_sizes, int n_in,
                              void* d_out, int out_size, void* d_ws, size_t ws_size,
                              hipStream_t stream) {
  (void)in_sizes; (void)n_in; (void)out_size; (void)ws_size;
  const float* x_mv = (const float*)d_in[0];
  const float* x_s  = (const float*)d_in[1];
  const float* c_mv = (const float*)d_in[2];
  const float* c_s  = (const float*)d_in[3];
  // params flattened jax-style: sorted group keys, then sorted leaf keys.
  // groups: 0 att_k, 1 att_o, 2 att_q, 3 att_v, 4 ca_k, 5 ca_o, 6 ca_q, 7 ca_v, 8 mlp1, 9 mlp2
  // leaves: 0 b_s, 1 w_mv, 2 w_mv2s, 3 w_s, 4 w_s2mv
  auto P = [&](int g, int j) -> const float* { return (const float*)d_in[4 + g * 5 + j]; };

  char* w = (char*)d_ws;
  size_t off = 0;
  auto alloc = [&](size_t bytes) -> char* {
    char* p = w + off; off += (bytes + 255) & ~(size_t)255; return p;
  };
  float* ln_mv  = (float*)alloc((size_t)TOK * 256 * 4);
  float* ln_s   = (float*)alloc((size_t)TOK * 64 * 4);
  float* cln_mv = (float*)alloc((size_t)TOK * 256 * 4);
  float* cln_s  = (float*)alloc((size_t)TOK * 64 * 4);
  float* q_mv   = (float*)alloc((size_t)TOK * 512 * 4);
  float* k_mv   = (float*)alloc((size_t)TOK * 512 * 4);
  float* v_mv   = (float*)alloc((size_t)TOK * 512 * 4);
  float* q_s    = (float*)alloc((size_t)TOK * 128 * 4);
  float* k_s    = (float*)alloc((size_t)TOK * 128 * 4);
  float* v_s    = (float*)alloc((size_t)TOK * 128 * 4);
  h16*   qh     = (h16*)  alloc((size_t)B_ * H_ * N_ * 96 * 2);
  h16*   kh     = (h16*)  alloc((size_t)B_ * H_ * M_ * 96 * 2);
  h16*   vh     = (h16*)  alloc((size_t)B_ * H_ * 96 * M_ * 2);  // feature-major
  float* o_mv   = (float*)alloc((size_t)TOK * 512 * 4);
  float* o_s    = (float*)alloc((size_t)TOK * 128 * 4);
  float* x1_mv  = (float*)alloc((size_t)TOK * 256 * 4);
  float* x1_s   = (float*)alloc((size_t)TOK * 64 * 4);
  float* g_mv   = (float*)alloc((size_t)TOK * 256 * 4);

  auto lin = [&](const float* imv, const float* is, int g,
                 const float* rmv, const float* rs,
                 float* omv, float* os, int IMV, int OMV, int IS, int OS) {
    equi_linear_kernel<<<TOK, 128, 0, stream>>>(imv, is,
        P(g,1), P(g,4), P(g,3), P(g,2), P(g,0), rmv, rs, omv, os, IMV, OMV, IS, OS);
  };
  const int packN = B_ * H_ * N_ * 96;
  const dim3 fgrid(N_ / (16 * QT * WPB), B_ * H_);

  // ---------------- Stage A: self-attention ----------------
  ln_kernel<<<TOK, 64, 0, stream>>>(x_mv, x_s, ln_mv, ln_s);
  lin(ln_mv, ln_s, 2, nullptr, nullptr, q_mv, q_s, 16, 32, 64, 128);   // att_q
  lin(ln_mv, ln_s, 0, nullptr, nullptr, k_mv, k_s, 16, 32, 64, 128);   // att_k
  lin(ln_mv, ln_s, 3, nullptr, nullptr, v_mv, v_s, 16, 32, 64, 128);   // att_v
  pack_kernel<<<(packN + 255) / 256, 256, 0, stream>>>(q_mv, q_s, qh, N_, 1);
  pack_kernel<<<(packN + 255) / 256, 256, 0, stream>>>(k_mv, k_s, kh, N_, 0);
  pack_v_kernel<<<(packN + 255) / 256, 256, 0, stream>>>(v_mv, v_s, vh, N_);
  flash_attn_kernel<<<fgrid, 32 * WPB, 0, stream>>>(qh, kh, vh, o_mv, o_s);
  lin(o_mv, o_s, 1, x_mv, x_s, x1_mv, x1_s, 32, 16, 128, 64);          // att_o + residual

  // ---------------- Stage B: cross-attention ----------------
  ln_kernel<<<TOK, 64, 0, stream>>>(x1_mv, x1_s, ln_mv, ln_s);
  ln_kernel<<<B_ * M_, 64, 0, stream>>>(c_mv, c_s, cln_mv, cln_s);
  lin(ln_mv,  ln_s,  6, nullptr, nullptr, q_mv, q_s, 16, 32, 64, 128); // ca_q
  lin(cln_mv, cln_s, 4, nullptr, nullptr, k_mv, k_s, 16, 32, 64, 128); // ca_k
  lin(cln_mv, cln_s, 7, nullptr, nullptr, v_mv, v_s, 16, 32, 64, 128); // ca_v
  pack_kernel<<<(packN + 255) / 256, 256, 0, stream>>>(q_mv, q_s, qh, N_, 1);
  pack_kernel<<<(packN + 255) / 256, 256, 0, stream>>>(k_mv, k_s, kh, M_, 0);
  pack_v_kernel<<<(packN + 255) / 256, 256, 0, stream>>>(v_mv, v_s, vh, M_);
  flash_attn_kernel<<<fgrid, 32 * WPB, 0, stream>>>(qh, kh, vh, o_mv, o_s);
  lin(o_mv, o_s, 5, x1_mv, x1_s, x1_mv, x1_s, 32, 16, 128, 64);        // ca_o + residual

  // ---------------- Stage C: geometric MLP ----------------
  ln_kernel<<<TOK, 64, 0, stream>>>(x1_mv, x1_s, ln_mv, ln_s);
  lin(ln_mv, ln_s, 8, nullptr, nullptr, q_mv, q_s, 16, 32, 64, 128);   // mlp1 (h_mv,h_s)
  geo_kernel<<<TOK, 256, 0, stream>>>(q_mv, g_mv);
  gelu_kernel<<<(TOK * 128 + 255) / 256, 256, 0, stream>>>(q_s, k_s, TOK * 128);
  float* out_mv = (float*)d_out;
  float* out_s  = out_mv + (size_t)TOK * 256;
  lin(g_mv, k_s, 9, x1_mv, x1_s, out_mv, out_s, 16, 16, 128, 64);      // mlp2 + residual
}
